// StepTopDownAttention_10196252360898
// MI455X (gfx1250) — matrix operand order repaired and checked by reading, not compiled, not run on hardware
//
#include <hip/hip_runtime.h>
#include <hip/hip_bf16.h>
#include <math.h>

// Sizes fixed by the reference
#define BB   16
#define NN   4096
#define DF   1024
#define DA   512

typedef __attribute__((ext_vector_type(16))) __bf16 v16bf;
typedef __attribute__((ext_vector_type(8)))  __bf16 v8bf;
typedef __attribute__((ext_vector_type(8)))  float  v8f;

__device__ __forceinline__ unsigned int f32_bf16(float f) {
  unsigned int u = __float_as_uint(f);
  u += 0x7FFFu + ((u >> 16) & 1u);          // round-to-nearest-even
  return (u >> 16) & 0xFFFFu;
}

// branchless sigmoid on v_exp_f32 + v_rcp_f32
__device__ __forceinline__ float fast_sigmoid(float x) {
  return __builtin_amdgcn_rcpf(1.0f + __expf(-x));
}

// ---------------------------------------------------------------------------
// 1) query-side projection: qp{1,2}[b][j] = query[b,:] . W{1,2}[1024: , j] + b{1,2}[j]
// ---------------------------------------------------------------------------
__global__ __launch_bounds__(512) void qproj_kernel(
    const float* __restrict__ query, const float* __restrict__ W1,
    const float* __restrict__ b1,    const float* __restrict__ W2,
    const float* __restrict__ b2,
    float* __restrict__ qp1, float* __restrict__ qp2)
{
  const int b = blockIdx.x, j = threadIdx.x;
  float a1 = b1[j], a2 = b2[j];
  const float* q = query + b * DF;
  for (int k = 0; k < DF; ++k) {
    const float qv = q[k];
    a1 = fmaf(qv, W1[(size_t)(DF + k) * DA + j], a1);
    a2 = fmaf(qv, W2[(size_t)(DF + k) * DA + j], a2);
  }
  qp1[b * DA + j] = a1;
  qp2[b * DA + j] = a2;
}

// ---------------------------------------------------------------------------
// 2) W key-half transpose + bf16 convert: WT[j][k] = bf16(W[k][j]), j<512, k<1024
// ---------------------------------------------------------------------------
__global__ __launch_bounds__(256) void wtrans_kernel(
    const float* __restrict__ W1, const float* __restrict__ W2,
    unsigned short* __restrict__ WT1, unsigned short* __restrict__ WT2)
{
  const int id = blockIdx.x * 256 + threadIdx.x;   // 0 .. 512*1024-1
  const int j = id >> 10, k = id & 1023;
  WT1[id] = (unsigned short)f32_bf16(W1[(size_t)k * DA + j]);
  WT2[id] = (unsigned short)f32_bf16(W2[(size_t)k * DA + j]);
}

// ---------------------------------------------------------------------------
// 3) fused score kernel: two bf16 WMMA GEMMs + gated-tanh + dot(w_a) -> score
//    block: 64 rows of flattened [B*N], 512 threads = 16 waves (4 Mw x 4 Nw)
//    register-prefetch double buffering: next k-chunk's global tile is loaded
//    into registers while the current chunk's WMMAs execute.
// ---------------------------------------------------------------------------
__global__ __launch_bounds__(512) void score_kernel(
    const float* __restrict__ key,
    const unsigned short* __restrict__ WT1,
    const unsigned short* __restrict__ WT2,
    const float* __restrict__ qp1, const float* __restrict__ qp2,
    const float* __restrict__ wa,
    float* __restrict__ score)
{
  // row stride 40 ushorts = 80 B: 16B-aligned rows + conflict-free 20-dword stride
  __shared__ __align__(16) unsigned short sK [64][40];
  __shared__ __align__(16) unsigned short sW1[256][40];
  __shared__ __align__(16) unsigned short sW2[256][40];
  __shared__ float sRed[4][64];

  const int tid  = threadIdx.x;
  const int lane = tid & 31;
  const int wid  = tid >> 5;
  const int mw   = wid & 3;        // wave's 16-row block
  const int nw   = wid >> 2;       // wave's 64-column block
  const int l16  = lane & 15;
  const int lh   = lane >> 4;      // lane half (K/M-group select)
  const int mbase = blockIdx.x * 64;
  const int b     = mbase >> 12;   // batch (N=4096 divides the tile)

  // staging roles
  const int krow = tid >> 3;            // key tile row 0..63
  const int kcol = (tid & 7) << 2;      // 0..28 step 4 (fp32 elems)
  const int wj   = tid >> 1;            // W tile row (column j) 0..255
  const int wkh  = (tid & 1) << 4;      // 0 / 16 (bf16 elems)

  const float* kgp = key + (size_t)(mbase + krow) * DF + kcol;

  float ssum[8];
  #pragma unroll
  for (int g = 0; g < 8; ++g) ssum[g] = 0.f;

  for (int jiter = 0; jiter < 2; ++jiter) {
    const int jbase = jiter << 8;       // 0 / 256
    v8f acc1[4], acc2[4];
    #pragma unroll
    for (int t = 0; t < 4; ++t)
      #pragma unroll
      for (int i = 0; i < 8; ++i) { acc1[t][i] = 0.f; acc2[t][i] = 0.f; }

    // ---- prefetch first tile of this jiter into registers
    float4 kv = *(const float4*)(kgp);
    size_t gw = (size_t)(jbase + wj) * DF + wkh;
    uint4 x0 = ((const uint4*)(WT1 + gw))[0];
    uint4 x1 = ((const uint4*)(WT1 + gw))[1];
    uint4 y0 = ((const uint4*)(WT2 + gw))[0];
    uint4 y1 = ((const uint4*)(WT2 + gw))[1];

    for (int k0 = 0; k0 < DF; k0 += 32) {
      __syncthreads();                  // previous chunk's fragment reads done
      { // commit prefetched tile to LDS (key fp32 -> bf16 in flight)
        const unsigned int p0 = f32_bf16(kv.x) | (f32_bf16(kv.y) << 16);
        const unsigned int p1 = f32_bf16(kv.z) | (f32_bf16(kv.w) << 16);
        *(unsigned int*)&sK[krow][kcol]     = p0;
        *(unsigned int*)&sK[krow][kcol + 2] = p1;
        *(uint4*)&sW1[wj][wkh]     = x0;
        *(uint4*)&sW1[wj][wkh + 8] = x1;
        *(uint4*)&sW2[wj][wkh]     = y0;
        *(uint4*)&sW2[wj][wkh + 8] = y1;
      }
      __syncthreads();

      // ---- prefetch next tile (overlaps with WMMA compute below)
      if (k0 + 32 < DF) {
        kv = *(const float4*)(kgp + k0 + 32);
        gw = (size_t)(jbase + wj) * DF + (k0 + 32) + wkh;
        x0 = ((const uint4*)(WT1 + gw))[0];
        x1 = ((const uint4*)(WT1 + gw))[1];
        y0 = ((const uint4*)(WT2 + gw))[0];
        y1 = ((const uint4*)(WT2 + gw))[1];
      }

      // A fragment: lane<16 -> M=l16, K 0..15; lane>=16 -> K 16..31
      const int arow = (mw << 4) + l16;
      const v8bf a0 = *(const v8bf*)&sK[arow][lh << 4];
      const v8bf a1 = *(const v8bf*)&sK[arow][(lh << 4) + 8];
      v16bf A;
      #pragma unroll
      for (int i = 0; i < 8; ++i) { A[i] = a0[i]; A[i + 8] = a1[i]; }

      #pragma unroll
      for (int t = 0; t < 4; ++t) {
        // B fragment: lane<16 -> col=l16, K 0..15; lane>=16 -> K 16..31
        const int bj = (nw << 6) + (t << 4) + l16;
        const v8bf w10 = *(const v8bf*)&sW1[bj][(lh << 4)];
        const v8bf w11 = *(const v8bf*)&sW1[bj][(lh << 4) + 8];
        const v8bf w20 = *(const v8bf*)&sW2[bj][(lh << 4)];
        const v8bf w21 = *(const v8bf*)&sW2[bj][(lh << 4) + 8];
        v16bf B1, B2;
        #pragma unroll
        for (int i = 0; i < 8; ++i) {
          B1[i] = w10[i]; B1[i + 8] = w11[i];
          B2[i] = w20[i]; B2[i + 8] = w21[i];
        }
        acc1[t] = __builtin_amdgcn_wmma_f32_16x16x32_bf16(
            false, A, false, B1, (short)0, acc1[t], false, false);
        acc2[t] = __builtin_amdgcn_wmma_f32_16x16x32_bf16(
            false, A, false, B2, (short)0, acc2[t], false, false);
      }
    }

    // fused epilogue: h = tanh(c1+qp1)*sigmoid(c2+qp2); ssum += h * wa[j]
    // C/D layout: vgpr g, lane -> row = g + 8*lh, col = l16
    #pragma unroll
    for (int t = 0; t < 4; ++t) {
      const int j = jbase + (nw << 6) + (t << 4) + l16;
      const float q1 = qp1[b * DA + j];
      const float q2 = qp2[b * DA + j];
      const float wv = wa[j];
      #pragma unroll
      for (int g = 0; g < 8; ++g) {
        const float c1 = acc1[t][g] + q1;
        const float c2 = acc2[t][g] + q2;
        // tanh(x) = 2*sigmoid(2x) - 1  (branchless, v_exp_f32 based)
        const float hv = (2.0f * fast_sigmoid(2.0f * c1) - 1.0f) * fast_sigmoid(c2);
        ssum[g] = fmaf(hv, wv, ssum[g]);
      }
    }
  }

  // deterministic reduce over the 16 columns held by each lane-half
  #pragma unroll
  for (int g = 0; g < 8; ++g) {
    float v = ssum[g];
    v += __shfl_xor(v, 1, 32);
    v += __shfl_xor(v, 2, 32);
    v += __shfl_xor(v, 4, 32);
    v += __shfl_xor(v, 8, 32);
    ssum[g] = v;
  }
  if (l16 == 0) {
    const int rb = (mw << 4) + (lh << 3);
    #pragma unroll
    for (int g = 0; g < 8; ++g) sRed[nw][rb + g] = ssum[g];
  }
  __syncthreads();
  if (tid < 64)
    score[mbase + tid] = sRed[0][tid] + sRed[1][tid] + sRed[2][tid] + sRed[3][tid];
}

// ---------------------------------------------------------------------------
// 4) masked softmax over N per batch
// ---------------------------------------------------------------------------
__global__ __launch_bounds__(1024) void softmax_kernel(
    const float* __restrict__ score, const int* __restrict__ mask,
    float* __restrict__ attn)
{
  __shared__ float red[1024];
  const int b = blockIdx.x, tid = threadIdx.x;
  float s[4];
  #pragma unroll
  for (int i = 0; i < 4; ++i) {
    const int n = tid + (i << 10);
    const float sc = score[b * NN + n];
    s[i] = (mask[b * NN + n] == 0) ? -1e18f : sc;
  }
  float mx = fmaxf(fmaxf(s[0], s[1]), fmaxf(s[2], s[3]));
  red[tid] = mx;
  __syncthreads();
  for (int st = 512; st > 0; st >>= 1) {
    if (tid < st) red[tid] = fmaxf(red[tid], red[tid + st]);
    __syncthreads();
  }
  mx = red[0];
  __syncthreads();
  float e[4]; float sum = 0.f;
  #pragma unroll
  for (int i = 0; i < 4; ++i) { e[i] = __expf(s[i] - mx); sum += e[i]; }
  red[tid] = sum;
  __syncthreads();
  for (int st = 512; st > 0; st >>= 1) {
    if (tid < st) red[tid] += red[tid + st];
    __syncthreads();
  }
  const float inv = 1.0f / red[0];
  #pragma unroll
  for (int i = 0; i < 4; ++i) attn[b * NN + tid + (i << 10)] = e[i] * inv;
}

// ---------------------------------------------------------------------------
// 5) context = attn @ value (bandwidth-bound), two-stage deterministic reduce
// ---------------------------------------------------------------------------
__global__ __launch_bounds__(256) void ctx_partial_kernel(
    const float* __restrict__ attn, const float* __restrict__ value,
    float* __restrict__ ctxPart)
{
  __shared__ float sr[128];
  const int b = blockIdx.z, nc = blockIdx.y, d0 = blockIdx.x << 7;
  const int td = threadIdx.x & 127, tg = threadIdx.x >> 7;
  const int n0 = (nc << 10) + (tg << 9);
  const float* ap = attn + b * NN;
  float acc = 0.f;
  #pragma unroll 4
  for (int i = 0; i < 512; ++i) {
    const int n = n0 + i;
    acc = fmaf(ap[n], value[((size_t)b * NN + n) * DF + d0 + td], acc);
  }
  if (tg) sr[td] = acc;
  __syncthreads();
  if (!tg) ctxPart[((size_t)(nc * BB + b)) * DF + d0 + td] = acc + sr[td];
}

__global__ __launch_bounds__(256) void ctx_reduce_kernel(
    const float* __restrict__ ctxPart, float* __restrict__ ctx)
{
  const int id = blockIdx.x * 256 + threadIdx.x;   // 0 .. 16383
  const int b = id >> 10, d = id & 1023;
  float s = 0.f;
  for (int nc = 0; nc < 4; ++nc) s += ctxPart[((size_t)(nc * BB + b)) * DF + d];
  ctx[id] = s;
}

// ---------------------------------------------------------------------------
extern "C" void kernel_launch(void* const* d_in, const int* in_sizes, int n_in,
                              void* d_out, int out_size, void* d_ws, size_t ws_size,
                              hipStream_t stream) {
  const float* query = (const float*)d_in[0];
  const float* key   = (const float*)d_in[1];
  const float* value = (const float*)d_in[2];
  const int*   mask  = (const int*)  d_in[3];
  const float* W1    = (const float*)d_in[4];
  const float* b1    = (const float*)d_in[5];
  const float* W2    = (const float*)d_in[6];
  const float* b2    = (const float*)d_in[7];
  const float* wa    = (const float*)d_in[8];

  char* ws = (char*)d_ws;
  float*          qp1   = (float*)(ws);                               // 32 KB
  float*          qp2   = (float*)(ws + 32768);                       // 32 KB
  unsigned short* WT1   = (unsigned short*)(ws + 65536);              // 1 MB
  unsigned short* WT2   = (unsigned short*)(ws + 65536 + 1048576);    // 1 MB
  float*          scoreB= (float*)(ws + 65536 + 2097152);             // 256 KB
  float*          ctxP  = (float*)(ws + 65536 + 2097152 + 262144);    // 256 KB

  float* attn = (float*)d_out;            // [16, 4096]
  float* ctx  = (float*)d_out + BB * NN;  // [16, 1024]

  qproj_kernel  <<<BB, 512, 0, stream>>>(query, W1, b1, W2, b2, qp1, qp2);
  wtrans_kernel <<<(DA * DF) / 256, 256, 0, stream>>>(W1, W2, WT1, WT2);
  score_kernel  <<<(BB * NN) / 64, 512, 0, stream>>>(key, WT1, WT2, qp1, qp2, wa, scoreB);
  softmax_kernel<<<BB, 1024, 0, stream>>>(scoreB, mask, attn);
  ctx_partial_kernel<<<dim3(DF / 128, 4, BB), 256, 0, stream>>>(attn, value, ctxP);
  ctx_reduce_kernel <<<(BB * DF) / 256, 256, 0, stream>>>(ctxP, ctx);
}
